// TileRRGEncoder_89644557402628
// MI455X (gfx1250) — matrix operand order repaired and checked by reading, not compiled, not run on hardware
//
#include <hip/hip_runtime.h>
#include <hip/hip_bf16.h>

// ---------------------------------------------------------------------------
// Types for CDNA5 WMMA
// ---------------------------------------------------------------------------
typedef __attribute__((ext_vector_type(16))) __bf16          v16bf;
typedef __attribute__((ext_vector_type(16))) unsigned short  v16us;
typedef __attribute__((ext_vector_type(8)))  float           v8f;
typedef __attribute__((ext_vector_type(4)))  float           f4v;
typedef __attribute__((ext_vector_type(4)))  int             v4i;

#if __has_builtin(__builtin_amdgcn_global_load_async_to_lds_b128) && \
    __has_builtin(__builtin_amdgcn_s_wait_asynccnt)
#define USE_ASYNC_LDS 1
typedef __attribute__((address_space(1))) v4i* gas_v4i_ptr;
typedef __attribute__((address_space(3))) v4i* las_v4i_ptr;
#endif

__device__ __forceinline__ unsigned short f2bf(float f) {
    unsigned int u = __builtin_bit_cast(unsigned int, f);
    unsigned int r = (u + 0x7FFFu + ((u >> 16) & 1u)) >> 16;
    return (unsigned short)r;
}

// ---------------------------------------------------------------------------
// Utility kernels
// ---------------------------------------------------------------------------
__global__ void zerok(float* __restrict__ p, long long n) {
    long long i = (long long)blockIdx.x * blockDim.x + threadIdx.x;
    if (i < n) p[i] = 0.0f;
}

__global__ void deg_kernel(const int* __restrict__ dst, float* __restrict__ deg, int E) {
    int i = blockIdx.x * blockDim.x + threadIdx.x;
    if (i < E) atomicAdd(&deg[dst[i]], 1.0f);
}

__global__ void invd_kernel(float* __restrict__ deg, int N) {
    int i = blockIdx.x * blockDim.x + threadIdx.x;
    if (i < N) deg[i] = 1.0f / fmaxf(deg[i], 1.0f);
}

// x0[i, 0:32] = [feat[i,0:13] | glob[i,0:16] | 0 0 0]
__global__ void pack_x0(const float* __restrict__ feat, const float* __restrict__ glob,
                        float* __restrict__ x0, int N) {
    int i = blockIdx.x * blockDim.x + threadIdx.x;
    if (i >= N * 32) return;
    int r = i >> 5, c = i & 31;
    float v = 0.0f;
    if (c < 13)      v = feat[r * 13 + c];
    else if (c < 29) v = glob[r * 16 + (c - 13)];
    x0[i] = v;
}

// Pack [Ws;Wn] (logical [2*Kpad x 64], zero-padded) directly into the WMMA
// B-fragment register layout:
//   frag index = kstep*4 + ct ; within a frag, lane L owns 16 contiguous bf16:
//     half = L>>4, m = L&15, element e -> k = kstep*32 + (e<8 ? half*8+e : 16+half*8+(e-8)),
//     n = ct*16 + m
// so every lane's fragment is one 32-byte contiguous block.
__global__ void pack_w_frag(const float* __restrict__ Ws, const float* __restrict__ Wn,
                            unsigned short* __restrict__ Wfrag, int Kin, int Kpad) {
    int i = blockIdx.x * blockDim.x + threadIdx.x;   // over 2*Kpad*64 elements
    if (i >= 2 * Kpad * 64) return;
    int e    = i & 15;
    int lane = (i >> 4) & 31;
    int fi   = i >> 9;                // frag index = kstep*4 + ct
    int ct   = fi & 3;
    int ks   = fi >> 2;
    int half = lane >> 4, m = lane & 15;
    int k = ks * 32 + (e < 8 ? half * 8 + e : 16 + half * 8 + (e - 8));
    int n = ct * 16 + m;
    float v = 0.0f;
    if (k < Kpad) { if (k < Kin) v = Ws[k * 64 + n]; }
    else          { int kk = k - Kpad; if (kk < Kin) v = Wn[kk * 64 + n]; }
    Wfrag[i] = f2bf(v);
}

// agg[dst[e], f:f+4] += X[src[e], f:f+4]   (one thread per (edge, 4 features))
__global__ void scatter_agg4(const float* __restrict__ X, int ldx, int lgK4,
                             const int* __restrict__ src, const int* __restrict__ dst,
                             float* __restrict__ agg, long long total) {
    long long i = (long long)blockIdx.x * blockDim.x + threadIdx.x;
    if (i >= total) return;
    int e  = (int)(i >> lgK4);
    int f4 = (int)(i & ((1 << lgK4) - 1)) << 2;
    int s = src[e], d = dst[e];
    f4v v = *(const f4v*)(X + (long long)s * ldx + f4);
    float* a = agg + (long long)d * 64 + f4;
    atomicAdd(a + 0, v[0]);
    atomicAdd(a + 1, v[1]);
    atomicAdd(a + 2, v[2]);
    atomicAdd(a + 3, v[3]);
}

// ---------------------------------------------------------------------------
// Fused SAGE layer GEMM:  Y = relu([X | agg*invd] @ Wcat + b)
//   Wave-per-16-row tile, 4 column tiles (H=64), fully-unrolled K loop,
//   v_wmma_f32_16x16x32_bf16. A built from float4 global loads; B fragments
//   pre-permuted so each lane loads 32 contiguous bytes from LDS (b128 x2).
//   Weight staging uses ASYNCcnt-tracked global_load_async_to_lds_b128.
// ---------------------------------------------------------------------------
template <int KPAD>
__global__ void __launch_bounds__(256)
sage_gemm(const float* __restrict__ X, int ldx,
          const float* __restrict__ agg, const float* __restrict__ invd,
          const unsigned short* __restrict__ Wfrag,
          const float* __restrict__ bias,
          float* __restrict__ Y, int ldy, int ntiles)
{
    constexpr int NSTEPS = (2 * KPAD) / 32;          // 2 (KPAD=32) or 4 (KPAD=64)
    constexpr int NFRAG_US = NSTEPS * 4 * 32 * 16;   // ushorts of fragment data
    __shared__ unsigned short sW[4 * 4 * 32 * 16];   // 16KB max
    __shared__ float sB[64];

    int t = threadIdx.x;
    // cooperative fragment staging into LDS (128-bit granularity)
    {
        const uint4* gsrc = (const uint4*)Wfrag;
        uint4* ldst = (uint4*)sW;
#ifdef USE_ASYNC_LDS
        for (int i = t; i < NFRAG_US / 8; i += 256) {
            const void* gv = (const void*)(gsrc + i);
            void*       lv = (void*)(ldst + i);
            __builtin_amdgcn_global_load_async_to_lds_b128(
                (gas_v4i_ptr)gv, (las_v4i_ptr)lv, 0, 0);
        }
        __builtin_amdgcn_s_wait_asynccnt(0);
#else
        for (int i = t; i < NFRAG_US / 8; i += 256) ldst[i] = gsrc[i];
#endif
    }
    if (t < 64) sB[t] = bias[t];
    __syncthreads();

    int wave = t >> 5;            // wave32
    int lane = t & 31;
    int tile = blockIdx.x * 8 + wave;
    if (tile >= ntiles) return;   // wave-uniform: EXEC all-ones for WMMA

    int m    = lane & 15;
    int half = lane >> 4;
    unsigned row = (unsigned)(tile * 16 + m);
    float inv = invd[row];

    // 32-bit offsets (max row*ldx = 12.8M floats, fits easily)
    const float* Xrow = X + (size_t)(row * (unsigned)ldx);
    const float* Arow = agg + (size_t)(row * 64u);

    v8f zero8 = {0.f, 0.f, 0.f, 0.f, 0.f, 0.f, 0.f, 0.f};
    v8f acc[4] = {zero8, zero8, zero8, zero8};

    #pragma unroll
    for (int s = 0; s < NSTEPS; ++s) {
        const bool isX = (s * 32) < KPAD;            // folds at compile time
        const float* base = isX ? (Xrow + s * 32) : (Arow + (s * 32 - KPAD));
        float sc = isX ? 1.0f : inv;

        // A fragment: elements 0..7 from k = half*8 + [0..7],
        //             elements 8..15 from k = 16 + half*8 + [0..7]
        const float* p0 = base + half * 8;
        const float* p1 = base + 16 + half * 8;
        f4v x0v = *(const f4v*)(p0);
        f4v x1v = *(const f4v*)(p0 + 4);
        f4v y0v = *(const f4v*)(p1);
        f4v y1v = *(const f4v*)(p1 + 4);
        v16us au;
        #pragma unroll
        for (int e = 0; e < 4; ++e) {
            au[e]      = f2bf(x0v[e] * sc);
            au[4 + e]  = f2bf(x1v[e] * sc);
            au[8 + e]  = f2bf(y0v[e] * sc);
            au[12 + e] = f2bf(y1v[e] * sc);
        }
        v16bf a = __builtin_bit_cast(v16bf, au);

        #pragma unroll
        for (int ct = 0; ct < 4; ++ct) {
            // lane's pre-permuted 32-byte fragment block
            const v16us* bp = (const v16us*)&sW[(((s * 4 + ct) * 32) + lane) * 16];
            v16bf b = __builtin_bit_cast(v16bf, *bp);
            acc[ct] = __builtin_amdgcn_wmma_f32_16x16x32_bf16(
                false, a, false, b, (short)0, acc[ct], false, false);
        }
    }

    // epilogue: bias + relu; C layout: VGPR r -> M = half*8 + r, N = m
    float* Yrow0 = Y + (size_t)((unsigned)(tile * 16 + half * 8) * (unsigned)ldy);
    #pragma unroll
    for (int ct = 0; ct < 4; ++ct) {
        int n = ct * 16 + m;
        float bv = sB[n];
        #pragma unroll
        for (int r = 0; r < 8; ++r) {
            float v = acc[ct][r] + bv;
            v = v > 0.0f ? v : 0.0f;
            Yrow0[(unsigned)(r * ldy) + n] = v;
        }
    }
}

// ---------------------------------------------------------------------------
// Graph pooling + tiny MLP
// ---------------------------------------------------------------------------
__global__ void gcount(const int* __restrict__ gid, float* __restrict__ cnt, int N) {
    int i = blockIdx.x * blockDim.x + threadIdx.x;
    if (i < N) atomicAdd(&cnt[gid[i]], 1.0f);
}

__global__ void gsum_kernel(const float* __restrict__ xjk, const float* __restrict__ glob,
                            const int* __restrict__ gid, float* __restrict__ gsum, int N) {
    long long i = (long long)blockIdx.x * blockDim.x + threadIdx.x;
    if (i >= (long long)N * 272) return;
    int r = (int)(i / 272);
    int c = (int)(i % 272);
    int g = gid[r];
    float v = (c < 256) ? xjk[(long long)r * 256 + c] : glob[(long long)r * 16 + (c - 256)];
    atomicAdd(&gsum[g * 272 + c], v);
}

__global__ void mlp1(const float* __restrict__ gsum, const float* __restrict__ cnt,
                     const float* __restrict__ W1, const float* __restrict__ bm1,
                     float* __restrict__ hidden) {
    int g = blockIdx.x, j = threadIdx.x;      // grid 64, block 64
    float invc = 1.0f / fmaxf(cnt[g], 1.0f);
    float s = bm1[j];
    for (int k = 0; k < 272; ++k)
        s += gsum[g * 272 + k] * invc * W1[k * 64 + j];
    hidden[g * 64 + j] = fmaxf(s, 0.0f);
}

__global__ void mlp2(const float* __restrict__ hidden, const float* __restrict__ W2,
                     const float* __restrict__ bm2, float* __restrict__ out) {
    int g = blockIdx.x, o = threadIdx.x;      // grid 64, block 256
    float s = bm2[o];
    for (int j = 0; j < 64; ++j)
        s += hidden[g * 64 + j] * W2[j * 256 + o];
    out[g * 256 + o] = s;
}

// ---------------------------------------------------------------------------
// Host orchestration
// ---------------------------------------------------------------------------
extern "C" void kernel_launch(void* const* d_in, const int* in_sizes, int n_in,
                              void* d_out, int out_size, void* d_ws, size_t ws_size,
                              hipStream_t stream) {
    const float* feat = (const float*)d_in[0];
    const float* glob = (const float*)d_in[1];
    const int*   src  = (const int*)d_in[2];
    const int*   dst  = (const int*)d_in[3];
    const int*   gid  = (const int*)d_in[4];
    const float* Ws[4] = {(const float*)d_in[5],  (const float*)d_in[8],
                          (const float*)d_in[11], (const float*)d_in[14]};
    const float* Wn[4] = {(const float*)d_in[6],  (const float*)d_in[9],
                          (const float*)d_in[12], (const float*)d_in[15]};
    const float* bb[4] = {(const float*)d_in[7],  (const float*)d_in[10],
                          (const float*)d_in[13], (const float*)d_in[16]};
    const float* W1  = (const float*)d_in[17];
    const float* bm1 = (const float*)d_in[18];
    const float* W2  = (const float*)d_in[19];
    const float* bm2 = (const float*)d_in[20];

    const int N = in_sizes[0] / 13;
    const int E = in_sizes[2];
    const int G = 64;

    // workspace carve-up (all chunks keep 16B alignment)
    char* ws = (char*)d_ws;
    float* x0   = (float*)ws;  ws += (size_t)N * 32 * sizeof(float);
    float* agg  = (float*)ws;  ws += (size_t)N * 64 * sizeof(float);
    float* invd = (float*)ws;  ws += (size_t)N * sizeof(float) + 48;  // pad to 16B
    unsigned short* wc[4];
    wc[0] = (unsigned short*)ws; ws += (size_t)64 * 64 * sizeof(unsigned short);
    for (int l = 1; l < 4; ++l) { wc[l] = (unsigned short*)ws; ws += (size_t)128 * 64 * sizeof(unsigned short); }
    float* gsum   = (float*)ws;  ws += (size_t)G * 272 * sizeof(float);
    float* gcnt   = (float*)ws;  ws += (size_t)G * sizeof(float);
    float* hidden = (float*)ws;  ws += (size_t)G * 64 * sizeof(float);

    float* xjk  = (float*)d_out;
    float* gemb = xjk + (size_t)N * 256;

    const int B = 256;
    auto blocks = [](long long n, int b) { return (unsigned)((n + b - 1) / b); };

    // degree -> reciprocal denominator (in-place in invd)
    zerok<<<blocks(N, B), B, 0, stream>>>(invd, N);
    deg_kernel<<<blocks(E, B), B, 0, stream>>>(dst, invd, E);
    invd_kernel<<<blocks(N, B), B, 0, stream>>>(invd, N);

    // input packing + fragment-layout weight packing
    pack_x0<<<blocks((long long)N * 32, B), B, 0, stream>>>(feat, glob, x0, N);
    pack_w_frag<<<blocks(2 * 32 * 64, B), B, 0, stream>>>(Ws[0], Wn[0], wc[0], 29, 32);
    for (int l = 1; l < 4; ++l)
        pack_w_frag<<<blocks(2 * 64 * 64, B), B, 0, stream>>>(Ws[l], Wn[l], wc[l], 64, 64);

    const int ntiles = N / 16;                 // 50000 = 16 * 3125 exactly
    const unsigned gemm_blocks = (unsigned)((ntiles + 7) / 8);

    for (int l = 0; l < 4; ++l) {
        const float* Xl; int ldx, lgK4;
        if (l == 0) { Xl = x0;                 ldx = 32;  lgK4 = 3; }
        else        { Xl = xjk + 64 * (l - 1); ldx = 256; lgK4 = 4; }

        zerok<<<blocks((long long)N * 64, B), B, 0, stream>>>(agg, (long long)N * 64);
        long long total = (long long)E << lgK4;
        scatter_agg4<<<blocks(total, B), B, 0, stream>>>(Xl, ldx, lgK4, src, dst, agg, total);

        if (l == 0)
            sage_gemm<32><<<gemm_blocks, 256, 0, stream>>>(Xl, ldx, agg, invd,
                                                           wc[l], bb[l],
                                                           xjk + 64 * l, 256, ntiles);
        else
            sage_gemm<64><<<gemm_blocks, 256, 0, stream>>>(Xl, ldx, agg, invd,
                                                           wc[l], bb[l],
                                                           xjk + 64 * l, 256, ntiles);
    }

    // graph pooling + MLP head
    zerok<<<blocks(G * 272, B), B, 0, stream>>>(gsum, G * 272);
    zerok<<<blocks(G, B), B, 0, stream>>>(gcnt, G);
    gcount<<<blocks(N, B), B, 0, stream>>>(gid, gcnt, N);
    gsum_kernel<<<blocks((long long)N * 272, B), B, 0, stream>>>(xjk, glob, gid, gsum, N);
    mlp1<<<G, 64, 0, stream>>>(gsum, gcnt, W1, bm1, hidden);
    mlp2<<<G, 256, 0, stream>>>(hidden, W2, bm2, gemb);
}
